// SPSAGEConv_69724499083322
// MI455X (gfx1250) — matrix operand order repaired and checked by reading, not compiled, not run on hardware
//
#include <hip/hip_runtime.h>
#include <hip/hip_bf16.h>

#define IN_FEATS  512
#define OUT_FEATS 512

typedef __attribute__((ext_vector_type(16))) __bf16 v16bf;
typedef __attribute__((ext_vector_type(8)))  float  v8f;

union BFrag { unsigned u[8]; uint4 q[2]; v16bf v; };

__device__ inline unsigned short f32_to_bf16(float f) {
    unsigned u = __float_as_uint(f);
    unsigned r = u + 0x7FFFu + ((u >> 16) & 1u);   // round-to-nearest-even
    return (unsigned short)(r >> 16);
}
__device__ inline float bf16_to_f32(unsigned short h) {
    return __uint_as_float(((unsigned)h) << 16);
}
__device__ inline unsigned pack2(unsigned short lo, unsigned short hi) {
    return ((unsigned)hi << 16) | (unsigned)lo;
}

// ---------------------------------------------------------------------------
// Pack W (row-major [OUT=512][IN=512] f32) into the WMMA B-operand register
// layout for v_wmma_f32_16x16x32_bf16, as hi/lo residual pairs.
// Flat dword index: idx = ks*8192 + ntile*256 + lane*8 + v   (lane-major, so a
// lane's whole 8-dword fragment is contiguous -> two global_load_b128)
//   lane: n = ntile*16 + (lane&15), khalf = lane>>4
//   dword v holds bf16x2 {K = ks*32 + 2v + 16*khalf, K+1}   (low half = K)
// ---------------------------------------------------------------------------
__global__ void pack_w_kernel(const float* __restrict__ W,
                              unsigned* __restrict__ Whi,
                              unsigned* __restrict__ Wlo)
{
    int idx  = blockIdx.x * blockDim.x + threadIdx.x;   // 0 .. 131071
    int v    = idx & 7;
    int lane = (idx >> 3) & 31;
    int nt   = (idx >> 8) & 31;
    int ks   = idx >> 13;
    int n    = nt * 16 + (lane & 15);
    int k    = ks * 32 + 2 * v + 16 * (lane >> 4);
    float a0 = W[n * IN_FEATS + k];
    float a1 = W[n * IN_FEATS + k + 1];
    unsigned short h0 = f32_to_bf16(a0), h1 = f32_to_bf16(a1);
    float r0 = a0 - bf16_to_f32(h0);
    float r1 = a1 - bf16_to_f32(h1);
    Whi[idx] = pack2(h0, h1);
    Wlo[idx] = pack2(f32_to_bf16(r0), f32_to_bf16(r1));
}

// ---------------------------------------------------------------------------
// C[M x 512] = A[M x 512] * W^T (+bias)  via bf16 split-WMMA, f32 accumulate.
// 256 threads (8 wave32) per WG; WG computes a 32-row x 512-col strip
// (2 M-tiles), wave w owns columns [64w, 64w+64) as 4 N-tiles.
// A slab (32x32 f32) is converted to packed bf16 hi/lo ONCE per K-step during
// cooperative staging; lanes then fetch fragments with ds_load_b128.
// LDS row stride 20 dwords: 16B-aligned and bank-conflict-free for b128 reads.
// ---------------------------------------------------------------------------
__global__ __launch_bounds__(256)
void gemm_bf16x3_kernel(const float* __restrict__ A,
                        const unsigned* __restrict__ Whi,
                        const unsigned* __restrict__ Wlo,
                        const float* __restrict__ bias,
                        float* __restrict__ C)
{
    __shared__ __align__(16) unsigned AhiL[32 * 20];
    __shared__ __align__(16) unsigned AloL[32 * 20];

    const int lane = threadIdx.x & 31;
    const int wave = threadIdx.x >> 5;
    const int half = lane >> 4;                     // 0 or 1
    const int l16  = lane & 15;                     // A row / B,C,D column
    const int row_base = blockIdx.x * 32;

    v8f acc[2][4] = {};                             // 2 M-tiles x 4 N-tiles x 8 f32

    // Fragment dword v lives at LDS pair-index kp = 8*half + v:
    //   v=0..3 -> kb0 + v,  v=4..7 -> kb1 + (v-4)
    const int kb0 = 8 * half;
    const int kb1 = 8 * half + 4;

    // staging decomposition: thread t loads float4 at rows m=t/8, k=(t%8)*4
    const int sm = threadIdx.x >> 3;
    const int sk = (threadIdx.x & 7) * 4;

    for (int ks = 0; ks < IN_FEATS / 32; ++ks) {
        __syncthreads();
        {   // cooperative load + one-time f32 -> bf16 hi/lo split (1024 f32)
            const float4 g = *(const float4*)(A + (size_t)(row_base + sm) * IN_FEATS
                                                + ks * 32 + sk);
            unsigned short h0 = f32_to_bf16(g.x), h1 = f32_to_bf16(g.y);
            unsigned short h2 = f32_to_bf16(g.z), h3 = f32_to_bf16(g.w);
            float r0 = g.x - bf16_to_f32(h0), r1 = g.y - bf16_to_f32(h1);
            float r2 = g.z - bf16_to_f32(h2), r3 = g.w - bf16_to_f32(h3);
            int kp = sk >> 1;                       // pair index (even)
            AhiL[sm * 20 + kp]     = pack2(h0, h1);
            AhiL[sm * 20 + kp + 1] = pack2(h2, h3);
            AloL[sm * 20 + kp]     = pack2(f32_to_bf16(r0), f32_to_bf16(r1));
            AloL[sm * 20 + kp + 1] = pack2(f32_to_bf16(r2), f32_to_bf16(r3));
        }
        __syncthreads();

        // Per-lane A fragments: 2 tiles x (hi,lo) x two ds_load_b128 each
        v16bf Ahi[2], Alo[2];
#pragma unroll
        for (int mt = 0; mt < 2; ++mt) {
            const unsigned* ph = &AhiL[(mt * 16 + l16) * 20];
            const unsigned* pl = &AloL[(mt * 16 + l16) * 20];
            BFrag fh, fl;
            fh.q[0] = *(const uint4*)(ph + kb0);
            fh.q[1] = *(const uint4*)(ph + kb1);
            fl.q[0] = *(const uint4*)(pl + kb0);
            fl.q[1] = *(const uint4*)(pl + kb1);
            Ahi[mt] = fh.v;
            Alo[mt] = fl.v;
        }

#pragma unroll
        for (int nt = 0; nt < 4; ++nt) {
            const int ntile = wave * 4 + nt;
            const uint4* bh = (const uint4*)(Whi + ks * 8192 + ntile * 256 + lane * 8);
            const uint4* bl = (const uint4*)(Wlo + ks * 8192 + ntile * 256 + lane * 8);
            BFrag fbh, fbl;
            fbh.q[0] = bh[0]; fbh.q[1] = bh[1];
            fbl.q[0] = bl[0]; fbl.q[1] = bl[1];
            v16bf Bhi = fbh.v, Blo = fbl.v;
#pragma unroll
            for (int mt = 0; mt < 2; ++mt) {
                // 3-term split: hi*hi + hi*lo + lo*hi  (error ~2^-17, fp32-class)
                acc[mt][nt] = __builtin_amdgcn_wmma_f32_16x16x32_bf16(
                    false, Ahi[mt], false, Bhi, (short)0, acc[mt][nt], false, false);
                acc[mt][nt] = __builtin_amdgcn_wmma_f32_16x16x32_bf16(
                    false, Ahi[mt], false, Blo, (short)0, acc[mt][nt], false, false);
                acc[mt][nt] = __builtin_amdgcn_wmma_f32_16x16x32_bf16(
                    false, Alo[mt], false, Bhi, (short)0, acc[mt][nt], false, false);
            }
        }
    }

    // Store: D vgpr j -> row = row_base + mt*16 + j + 8*half, col = ntile*16 + l16
#pragma unroll
    for (int nt = 0; nt < 4; ++nt) {
        int n = (wave * 4 + nt) * 16 + l16;
        float b = bias ? bias[n] : 0.0f;
#pragma unroll
        for (int mt = 0; mt < 2; ++mt) {
#pragma unroll
            for (int j = 0; j < 8; ++j) {
                int row = row_base + mt * 16 + j + 8 * half;
                C[(size_t)row * OUT_FEATS + n] = acc[mt][nt][j] + b;
            }
        }
    }
}

// ---------------------------------------------------------------------------
// One wave per edge: acc[dst] += rows[src]; cnt[dst] += 1  (f32 atomics in L2)
// ---------------------------------------------------------------------------
__global__ __launch_bounds__(256)
void scatter_accum_kernel(const float* __restrict__ rows,
                          const int* __restrict__ e_src,
                          const int* __restrict__ e_dst,
                          float* __restrict__ acc,
                          float* __restrict__ cnt,
                          int n_edges)
{
    int lane = threadIdx.x & 31;
    int e = (blockIdx.x * blockDim.x + threadIdx.x) >> 5;
    if (e >= n_edges) return;
    int s = e_src[e];
    int d = e_dst[e];
    const float4* srow = (const float4*)(rows + (size_t)s * OUT_FEATS);
    float* arow = acc + (size_t)d * OUT_FEATS;
#pragma unroll
    for (int i = 0; i < 4; ++i) {
        int c = lane + i * 32;                      // 128 float4 per row
        float4 v = srow[c];
        atomicAdd(arow + c * 4 + 0, v.x);
        atomicAdd(arow + c * 4 + 1, v.y);
        atomicAdd(arow + c * 4 + 2, v.z);
        atomicAdd(arow + c * 4 + 3, v.w);
    }
    if (lane == 0) atomicAdd(cnt + d, 1.0f);
}

__global__ void div_rows_kernel(float* __restrict__ data,
                                const float* __restrict__ cnt, int rows)
{
    int idx = blockIdx.x * blockDim.x + threadIdx.x;
    if (idx >= rows * OUT_FEATS) return;
    float c = cnt[idx >> 9];
    data[idx] = data[idx] / fmaxf(c, 1.0f);
}

__global__ void finalize_kernel(float* __restrict__ out,
                                const float* __restrict__ acc,
                                const float* __restrict__ cnt, int rows)
{
    int idx = blockIdx.x * blockDim.x + threadIdx.x;
    if (idx >= rows * OUT_FEATS) return;
    float c = cnt[idx >> 9];
    out[idx] = out[idx] + acc[idx] / fmaxf(c, 1.0f);
}

// ---------------------------------------------------------------------------
extern "C" void kernel_launch(void* const* d_in, const int* in_sizes, int n_in,
                              void* d_out, int out_size, void* d_ws, size_t ws_size,
                              hipStream_t stream)
{
    const float* feat    = (const float*)d_in[0];
    const float* W_self  = (const float*)d_in[1];
    const float* b_self  = (const float*)d_in[2];
    const float* W_neigh = (const float*)d_in[3];
    const int*   e0_src  = (const int*)d_in[4];
    const int*   e0_dst  = (const int*)d_in[5];
    const int*   e1_src  = (const int*)d_in[6];
    const int*   e1_dst  = (const int*)d_in[7];

    const int n_e0    = in_sizes[4];
    const int n_e1    = in_sizes[6];
    const int num_dst = out_size / OUT_FEATS;                    // 8192
    const int num_src = in_sizes[0] / IN_FEATS - num_dst;        // 131072
    const int num_vir = 32768;                                   // fixed by reference setup

    // workspace carve-up
    char* ws = (char*)d_ws;
    float* h_src   = (float*)ws;  ws += (size_t)num_src * OUT_FEATS * 4;   // 268 MB
    char*  zbeg    = ws;
    float* vir_acc = (float*)ws;  ws += (size_t)num_vir * OUT_FEATS * 4;   // 64 MB
    float* vir_cnt = (float*)ws;  ws += (size_t)num_vir * 4;
    float* dst_acc = (float*)ws;  ws += (size_t)num_dst * OUT_FEATS * 4;   // 16 MB
    float* dst_cnt = (float*)ws;  ws += (size_t)num_dst * 4;
    char*  zend    = ws;
    unsigned* Whi_s = (unsigned*)ws;  ws += (size_t)(IN_FEATS * OUT_FEATS / 2) * 4;
    unsigned* Wlo_s = (unsigned*)ws;  ws += (size_t)(IN_FEATS * OUT_FEATS / 2) * 4;
    unsigned* Whi_n = (unsigned*)ws;  ws += (size_t)(IN_FEATS * OUT_FEATS / 2) * 4;
    unsigned* Wlo_n = (unsigned*)ws;  ws += (size_t)(IN_FEATS * OUT_FEATS / 2) * 4;

    float* out = (float*)d_out;

    // zero all accumulators in one capture-safe memset
    hipMemsetAsync(zbeg, 0, (size_t)(zend - zbeg), stream);

    // pack both weight matrices into WMMA B layout (hi + lo residual)
    {
        int nDw = IN_FEATS * OUT_FEATS / 2;        // 131072 dwords
        pack_w_kernel<<<nDw / 256, 256, 0, stream>>>(W_self,  Whi_s, Wlo_s);
        pack_w_kernel<<<nDw / 256, 256, 0, stream>>>(W_neigh, Whi_n, Wlo_n);
    }

    // h_dst = feat[:num_dst] @ W_self^T + b_self   -> d_out
    gemm_bf16x3_kernel<<<num_dst / 32, 256, 0, stream>>>(
        feat, Whi_s, Wlo_s, b_self, out);

    // h_src = feat[num_dst:] @ W_neigh^T           -> ws
    gemm_bf16x3_kernel<<<num_src / 32, 256, 0, stream>>>(
        feat + (size_t)num_dst * IN_FEATS, Whi_n, Wlo_n, nullptr, h_src);

    // h_vir = seg_mean(h_src[e0_src], e0_dst)
    scatter_accum_kernel<<<(n_e0 * 32 + 255) / 256, 256, 0, stream>>>(
        h_src, e0_src, e0_dst, vir_acc, vir_cnt, n_e0);
    div_rows_kernel<<<(num_vir * OUT_FEATS) / 256, 256, 0, stream>>>(
        vir_acc, vir_cnt, num_vir);

    // h_neigh accumulation = seg_sum(h_vir[e1_src], e1_dst)
    scatter_accum_kernel<<<(n_e1 * 32 + 255) / 256, 256, 0, stream>>>(
        vir_acc, e1_src, e1_dst, dst_acc, dst_cnt, n_e1);

    // out = h_dst + h_neigh (divide fused here)
    finalize_kernel<<<(num_dst * OUT_FEATS) / 256, 256, 0, stream>>>(
        out, dst_acc, dst_cnt, num_dst);
}
